// SDPAttention_26310969655637
// MI455X (gfx1250) — matrix operand order repaired and checked by reading, not compiled, not run on hardware
//
#include <hip/hip_runtime.h>
#include <hip/hip_bf16.h>
#include <math.h>

#define HEADS 16
#define DHID  64
#define EMB   1024
#define BB    4
#define QL    2048
#define KLEN  1024

typedef __bf16 bf16;
typedef __attribute__((ext_vector_type(16))) __bf16 v16bf;
typedef __attribute__((ext_vector_type(8)))  __bf16 v8bf;
typedef __attribute__((ext_vector_type(8)))  float  v8f;
typedef __attribute__((ext_vector_type(4)))  unsigned int u32x4;
typedef __attribute__((ext_vector_type(8)))  int i32x8;
typedef __attribute__((ext_vector_type(4)))  int i32x4;

#define HAVE_TDM __has_builtin(__builtin_amdgcn_tensor_load_to_lds)

static __device__ __forceinline__ v8f zero8() {
    v8f z = {0.f,0.f,0.f,0.f,0.f,0.f,0.f,0.f};
    return z;
}

static __device__ __forceinline__ v8f wmma_bf16(v16bf a, v16bf b, v8f c) {
    return __builtin_amdgcn_wmma_f32_16x16x32_bf16(false, a, false, b,
                                                   (short)0, c, false, false);
}

// A-layout fragment: elements 0..7 at *lo (k=16g+e), 8..15 at *hi (k=16g+8+e)
static __device__ __forceinline__ v16bf frag_from2(const bf16* lo, const bf16* hi) {
    v8bf l = *(const v8bf*)lo;
    v8bf h = *(const v8bf*)hi;
    v16bf r;
#pragma unroll
    for (int i = 0; i < 8; ++i) { r[i] = l[i]; r[i + 8] = h[i]; }
    return r;
}

// B-layout fragment: 16 contiguous bf16 (rows k = 16g..16g+15 of lane's column)
static __device__ __forceinline__ v16bf frag16(const bf16* p) {
    v8bf l = *(const v8bf*)p;
    v8bf h = *(const v8bf*)(p + 8);
    v16bf r;
#pragma unroll
    for (int i = 0; i < 8; ++i) { r[i] = l[i]; r[i + 8] = h[i]; }
    return r;
}

// --- Tensor Data Mover: async 2-D bf16 tile load, global -> LDS --------------
// pad flags: pad_enable(bit20) | pad_interval<<22 | pad_amount<<25
// K tile rows: 128B data + 16B pad  -> interval code 4 (32 dwords), amount 3 (4 dwords)
#define PAD_K ((1u << 20) | (4u << 22) | (3u << 25))
// V^T/B^T rows: 64B data + 16B pad -> interval code 3 (16 dwords), amount 3 (4 dwords)
#define PAD_V ((1u << 20) | (3u << 22) | (3u << 25))

static __device__ __forceinline__ void tdm_load_2d(bf16* lds_ptr,
                                                   const bf16* gptr,
                                                   unsigned int tensor_d0,
                                                   unsigned int tensor_d1,
                                                   unsigned int tile_d0,
                                                   unsigned int tile_d1,
                                                   unsigned int stride0,
                                                   unsigned int padflags) {
#if HAVE_TDM
    unsigned int lds_off = (unsigned int)(uintptr_t)lds_ptr; // LDS byte offset
    unsigned long long ga = (unsigned long long)(uintptr_t)gptr;
    u32x4 g0;
    g0[0] = 1u;                                    // count=1, user mode
    g0[1] = lds_off;                               // lds_addr
    g0[2] = (unsigned int)ga;                      // global_addr[31:0]
    g0[3] = (unsigned int)(ga >> 32) | (2u << 30); // global_addr[56:32] | type=2
    i32x8 g1;
    g1[0] = (int)((1u << 16) | padflags);          // data_size=2B + pad config
    g1[1] = (int)(tensor_d0 << 16);                // tensor_dim0[15:0]
    g1[2] = (int)((tensor_d0 >> 16) | (tensor_d1 << 16));
    g1[3] = (int)((tensor_d1 >> 16) | (tile_d0 << 16));
    g1[4] = (int)tile_d1;                          // tile_dim1 (tile_dim2 = 0)
    g1[5] = (int)stride0;                          // tensor_dim0_stride[31:0]
    g1[6] = 0;
    g1[7] = 0;
    i32x4 z4 = {0, 0, 0, 0};
    i32x8 z8 = {0, 0, 0, 0, 0, 0, 0, 0};
    __builtin_amdgcn_tensor_load_to_lds(g0, g1, z4, z4, z8, 0);
#else
    // fallback: wave-strided copy into the padded LDS layout
    const int lane = threadIdx.x & 31;
    const unsigned int ldsStride = tile_d0 + 8u;   // +16B pad in bf16 units
    for (unsigned int r = 0; r < tile_d1; ++r)
        for (unsigned int c = lane; c < tile_d0; c += 32)
            lds_ptr[r * ldsStride + c] = gptr[(size_t)r * stride0 + c];
    (void)tensor_d0; (void)tensor_d1; (void)padflags;
#endif
}

static __device__ __forceinline__ void tdm_wait(int more_in_flight) {
#if HAVE_TDM
    if (more_in_flight) __builtin_amdgcn_s_wait_tensorcnt(2);
    else                __builtin_amdgcn_s_wait_tensorcnt(0);
#else
    (void)more_in_flight;
#endif
}

// ---------------------------------------------------------------------------
// Kernel 0: one-time f32->bf16 conversion of K (head-major) and V (head-major,
// transposed to [n][key]) so the hot kernels move raw bf16 tiles via TDM.
// ---------------------------------------------------------------------------
__global__ __launch_bounds__(256)
void convert_kernel(const float* __restrict__ k, const float* __restrict__ v,
                    bf16* __restrict__ k_ws, bf16* __restrict__ vt_ws) {
    __shared__ float lds[64][65];
    const int t = threadIdx.x;
    const int kt = blockIdx.x, h = blockIdx.y, b = blockIdx.z;
    const size_t gbase = (size_t)b * KLEN * EMB + (size_t)(kt * 64) * EMB + h * DHID;
    bf16* kw = k_ws + (((size_t)b * HEADS + h) * KLEN + kt * 64) * DHID;
#pragma unroll
    for (int i = 0; i < 16; ++i) {
        int idx = i * 256 + t;
        int key = idx >> 6, d = idx & 63;
        float kv = k[gbase + (size_t)key * EMB + d];
        float vv = v[gbase + (size_t)key * EMB + d];
        kw[(size_t)key * DHID + d] = (bf16)kv;
        lds[key][d] = vv;
    }
    __syncthreads();
    bf16* vw = vt_ws + (((size_t)b * EMB + h * DHID) * KLEN + kt * 64);
#pragma unroll
    for (int i = 0; i < 16; ++i) {
        int idx = i * 256 + t;
        int d = idx >> 6, key = idx & 63;
        vw[(size_t)d * KLEN + key] = (bf16)lds[key][d];
    }
}

// ---------------------------------------------------------------------------
// Kernel 1: flash attention.  128 thr (4 waves) per (b, h, 64-q tile).
// K/V^T 32-key chunks DMA'd by the TDM (double-buffered, padded LDS rows).
// ---------------------------------------------------------------------------
__global__ __launch_bounds__(128)
void attn_kernel(const float* __restrict__ q, const bf16* __restrict__ k_ws,
                 const bf16* __restrict__ vt_ws, const int* __restrict__ mask,
                 float* __restrict__ attn) {
    __shared__ bf16 lds_q[64][72];
    __shared__ bf16 lds_k[2][32][72];   // 128B rows + 16B TDM pad
    __shared__ bf16 lds_vt[2][64][40];  // 64B rows + 16B TDM pad
    __shared__ bf16 lds_p[4][16][32];

    const int t    = threadIdx.x;
    const int wave = t >> 5;
    const int lane = t & 31;
    const int nl   = lane & 15;
    const int g    = lane >> 4;

    const int qtile = blockIdx.x;
    const int h     = blockIdx.y;
    const int b     = blockIdx.z;

    const size_t qbase = (size_t)b * QL * EMB + (size_t)(qtile * 64) * EMB + h * DHID;
    const size_t mbase = (size_t)b * QL * KLEN + (size_t)(qtile * 64) * KLEN;
    const bf16* kt_base = k_ws + ((size_t)b * HEADS + h) * KLEN * DHID;
    const bf16* vt_base = vt_ws + ((size_t)b * EMB + h * DHID) * KLEN;

    // kick off chunk 0 DMA, then stage Q tile (f32 -> bf16) while it flies
    if (wave == 0) {
        tdm_load_2d(&lds_k[0][0][0], kt_base, DHID, KLEN,
                    DHID, 32, DHID, PAD_K);
        tdm_load_2d(&lds_vt[0][0][0], vt_base, KLEN, DHID,
                    32, DHID, KLEN, PAD_V);
    }
#pragma unroll
    for (int i = 0; i < 32; ++i) {
        int idx = i * 128 + t;
        int r = idx >> 6, c = idx & 63;
        lds_q[r][c] = (bf16)q[qbase + (size_t)r * EMB + c];
    }
    __syncthreads();

    const int qm = wave * 16 + nl;
    v16bf qa0 = frag_from2(&lds_q[qm][16 * g],      &lds_q[qm][8 + 16 * g]);
    v16bf qa1 = frag_from2(&lds_q[qm][32 + 16 * g], &lds_q[qm][40 + 16 * g]);

    v8f o0 = zero8(), o1 = zero8(), o2 = zero8(), o3 = zero8();
    float rm[8], rl[8];
#pragma unroll
    for (int e = 0; e < 8; ++e) { rm[e] = -3.0e38f; rl[e] = 0.f; }

    const float scale = 0.03125f; // 1/sqrt(1024)

    for (int kc = 0; kc < KLEN; kc += 32) {
        const int buf = (kc >> 5) & 1;
        if (wave == 0) {
            if (kc + 32 < KLEN) {   // prefetch next chunk into other buffer
                tdm_load_2d(&lds_k[buf ^ 1][0][0],
                            kt_base + (size_t)(kc + 32) * DHID,
                            DHID, KLEN, DHID, 32, DHID, PAD_K);
                tdm_load_2d(&lds_vt[buf ^ 1][0][0],
                            vt_base + (kc + 32),
                            KLEN, DHID, 32, DHID, KLEN, PAD_V);
                tdm_wait(1);        // retire current chunk, leave 2 in flight
            } else {
                tdm_wait(0);
            }
        }
        __syncthreads();

        // S = Q @ K^T for two 16-key column blocks
        v8f s0 = zero8(), s1 = zero8();
        s0 = wmma_bf16(qa0, frag16(&lds_k[buf][nl][16 * g]),       s0);
        s0 = wmma_bf16(qa1, frag16(&lds_k[buf][nl][32 + 16 * g]),  s0);
        s1 = wmma_bf16(qa0, frag16(&lds_k[buf][16 + nl][16 * g]),      s1);
        s1 = wmma_bf16(qa1, frag16(&lds_k[buf][16 + nl][32 + 16 * g]), s1);

        // mask + online softmax (rows span the 16-lane halves in C layout)
#pragma unroll
        for (int e = 0; e < 8; ++e) {
            int qrow = wave * 16 + e + 8 * g;
            const int* mp = mask + mbase + (size_t)qrow * KLEN + kc;
            float a0 = (mp[nl]      != 0) ? s0[e] * scale : -1.0e30f;
            float a1 = (mp[16 + nl] != 0) ? s1[e] * scale : -1.0e30f;
            float mx = fmaxf(a0, a1);
            mx = fmaxf(mx, __shfl_xor(mx, 1, 32));
            mx = fmaxf(mx, __shfl_xor(mx, 2, 32));
            mx = fmaxf(mx, __shfl_xor(mx, 4, 32));
            mx = fmaxf(mx, __shfl_xor(mx, 8, 32));
            float mnew = fmaxf(rm[e], mx);
            float corr = __expf(rm[e] - mnew);
            float p0 = __expf(a0 - mnew);
            float p1 = __expf(a1 - mnew);
            float rs = p0 + p1;
            rs += __shfl_xor(rs, 1, 32);
            rs += __shfl_xor(rs, 2, 32);
            rs += __shfl_xor(rs, 4, 32);
            rs += __shfl_xor(rs, 8, 32);
            rl[e] = rl[e] * corr + rs;
            rm[e] = mnew;
            o0[e] *= corr; o1[e] *= corr; o2[e] *= corr; o3[e] *= corr;
            int m = e + 8 * g;
            lds_p[wave][m][nl]      = (bf16)p0;
            lds_p[wave][m][16 + nl] = (bf16)p1;
        }
        // per-wave P round-trip: same-wave DS ordering, no barrier needed
        v16bf pa = frag_from2(&lds_p[wave][nl][16 * g], &lds_p[wave][nl][8 + 16 * g]);
        o0 = wmma_bf16(pa, frag16(&lds_vt[buf][0  + nl][16 * g]), o0);
        o1 = wmma_bf16(pa, frag16(&lds_vt[buf][16 + nl][16 * g]), o1);
        o2 = wmma_bf16(pa, frag16(&lds_vt[buf][32 + nl][16 * g]), o2);
        o3 = wmma_bf16(pa, frag16(&lds_vt[buf][48 + nl][16 * g]), o3);
        __syncthreads();   // all reads of buf done before TDM refills it
    }

    float* ap = attn + (size_t)b * QL * EMB + (size_t)(qtile * 64) * EMB + h * DHID;
#pragma unroll
    for (int e = 0; e < 8; ++e) {
        int qrow = wave * 16 + e + 8 * g;
        float inv = 1.0f / rl[e];
        ap[(size_t)qrow * EMB + 0  + nl] = o0[e] * inv;
        ap[(size_t)qrow * EMB + 16 + nl] = o1[e] * inv;
        ap[(size_t)qrow * EMB + 32 + nl] = o2[e] * inv;
        ap[(size_t)qrow * EMB + 48 + nl] = o3[e] * inv;
    }
}

// ---------------------------------------------------------------------------
// Kernel 2: softmax over model dim (1024) per (b,q) row -> `out` output
// ---------------------------------------------------------------------------
__global__ __launch_bounds__(256)
void softmax_kernel(const float* __restrict__ attn, float* __restrict__ out) {
    __shared__ float red[8];
    const int row = blockIdx.x;
    const int t = threadIdx.x;
    const float* src = attn + (size_t)row * EMB;
    float* dst = out + (size_t)row * EMB;

    float x[4];
    float mx = -3.0e38f;
#pragma unroll
    for (int i = 0; i < 4; ++i) { x[i] = src[i * 256 + t]; mx = fmaxf(mx, x[i]); }
    for (int off = 1; off < 32; off <<= 1) mx = fmaxf(mx, __shfl_xor(mx, off, 32));
    if ((t & 31) == 0) red[t >> 5] = mx;
    __syncthreads();
    float bmx = red[0];
#pragma unroll
    for (int i = 1; i < 8; ++i) bmx = fmaxf(bmx, red[i]);
    __syncthreads();

    float s = 0.f;
#pragma unroll
    for (int i = 0; i < 4; ++i) { x[i] = __expf(x[i] - bmx); s += x[i]; }
    for (int off = 1; off < 32; off <<= 1) s += __shfl_xor(s, off, 32);
    if ((t & 31) == 0) red[t >> 5] = s;
    __syncthreads();
    float bs = 0.f;
#pragma unroll
    for (int i = 0; i < 8; ++i) bs += red[i];
    float inv = 1.0f / bs;
#pragma unroll
    for (int i = 0; i < 4; ++i) dst[i * 256 + t] = x[i] * inv;
}

// ---------------------------------------------------------------------------
// Kernel 3: v2 = out @ v.  B^T tiles DMA'd by TDM from pre-transposed bf16 v.
// ---------------------------------------------------------------------------
__global__ __launch_bounds__(128)
void gemm_kernel(const float* __restrict__ outp, const bf16* __restrict__ vt_ws,
                 float* __restrict__ v2) {
    __shared__ bf16 lds_a[64][40];
    __shared__ bf16 lds_bt[2][64][40];

    const int t    = threadIdx.x;
    const int wave = t >> 5;
    const int lane = t & 31;
    const int nl   = lane & 15;
    const int g    = lane >> 4;

    const int ntile = blockIdx.x;
    const int mtile = blockIdx.y;
    const int b     = blockIdx.z;

    const float* A  = outp + (size_t)b * QL * EMB + (size_t)(mtile * 64) * EMB;
    const bf16* bt_base = vt_ws + ((size_t)b * EMB + ntile * 64) * KLEN;

    v8f o0 = zero8(), o1 = zero8(), o2 = zero8(), o3 = zero8();

    if (wave == 0)
        tdm_load_2d(&lds_bt[0][0][0], bt_base,
                    KLEN, DHID, 32, DHID, KLEN, PAD_V);

    for (int kc = 0; kc < KLEN; kc += 32) {
        const int buf = (kc >> 5) & 1;
        __syncthreads();   // previous compute done before refilling buffers
        if (wave == 0) {
            if (kc + 32 < KLEN) {
                tdm_load_2d(&lds_bt[buf ^ 1][0][0],
                            bt_base + (kc + 32),
                            KLEN, DHID, 32, DHID, KLEN, PAD_V);
                tdm_wait(1);
            } else {
                tdm_wait(0);
            }
        }
        // stage A chunk (f32 -> bf16)
#pragma unroll
        for (int i = 0; i < 16; ++i) {
            int idx = i * 128 + t;
            int r = idx >> 5, c = idx & 31;
            lds_a[r][c] = (bf16)A[(size_t)r * EMB + kc + c];
        }
        __syncthreads();
        const int am = wave * 16 + nl;
        v16bf a = frag_from2(&lds_a[am][16 * g], &lds_a[am][8 + 16 * g]);
        o0 = wmma_bf16(a, frag16(&lds_bt[buf][0  + nl][16 * g]), o0);
        o1 = wmma_bf16(a, frag16(&lds_bt[buf][16 + nl][16 * g]), o1);
        o2 = wmma_bf16(a, frag16(&lds_bt[buf][32 + nl][16 * g]), o2);
        o3 = wmma_bf16(a, frag16(&lds_bt[buf][48 + nl][16 * g]), o3);
    }

    float* C = v2 + (size_t)b * QL * EMB + (size_t)(mtile * 64) * EMB + ntile * 64;
#pragma unroll
    for (int e = 0; e < 8; ++e) {
        int m = wave * 16 + e + 8 * g;
        C[(size_t)m * EMB + 0  + nl] = o0[e];
        C[(size_t)m * EMB + 16 + nl] = o1[e];
        C[(size_t)m * EMB + 32 + nl] = o2[e];
        C[(size_t)m * EMB + 48 + nl] = o3[e];
    }
}

// ---------------------------------------------------------------------------
extern "C" void kernel_launch(void* const* d_in, const int* in_sizes, int n_in,
                              void* d_out, int out_size, void* d_ws, size_t ws_size,
                              hipStream_t stream) {
    const float* q    = (const float*)d_in[0];
    const float* k    = (const float*)d_in[1];
    const float* v    = (const float*)d_in[2];
    const int*   mask = (const int*)d_in[3];

    float* v2   = (float*)d_out;
    float* out2 = (float*)d_out + (size_t)BB * QL * EMB;

    bf16*  k_ws  = (bf16*)d_ws;                              // 8 MB
    bf16*  vt_ws = k_ws + (size_t)BB * HEADS * KLEN * DHID;  // 8 MB
    float* attn  = (float*)(vt_ws + (size_t)BB * EMB * KLEN);// 32 MB

    convert_kernel<<<dim3(KLEN / 64, HEADS, BB), 256, 0, stream>>>(k, v, k_ws, vt_ws);
    attn_kernel<<<dim3(QL / 64, HEADS, BB), 128, 0, stream>>>(q, k_ws, vt_ws, mask, attn);
    softmax_kernel<<<BB * QL, 256, 0, stream>>>(attn, out2);
    gemm_kernel<<<dim3(EMB / 64, QL / 64, BB), 128, 0, stream>>>(out2, vt_ws, v2);
}
